// AttentionFFB_26310969655495
// MI455X (gfx1250) — matrix operand order repaired and checked by reading, not compiled, not run on hardware
//
#include <hip/hip_runtime.h>
#include <hip/hip_bf16.h>

// ---------------------------------------------------------------------------
// CDNA5 (gfx1250) gated attention, bf16 WMMA with fp32 accumulation.
// Shapes: b0=1, b1=128, n=256, c=256, h=8, d=32.
// ---------------------------------------------------------------------------

typedef __attribute__((ext_vector_type(16))) __bf16 v16bf;
typedef __attribute__((ext_vector_type(8)))  __bf16 v8bf;
typedef __attribute__((ext_vector_type(8)))  float  v8f;
typedef __attribute__((ext_vector_type(4)))  float  v4f;

#define ATTN_SCALE 0.17677669529663687f   /* 1/sqrt(32) */
#define MASK_NEG   1.0e9f

__device__ __forceinline__ float fast_rcp(float x) {
    return __builtin_amdgcn_rcpf(x);
}

// ---------------------------------------------------------------------------
// Generic GEMM: Y[M,N] = X[M,K] * W[K,N] (+ bias[N]) (opt sigmoid epilogue)
// One wave computes one 16x16 C tile; K fully unrolled (K/32 static WMMAs).
// A-frag (16x32, 16-bit): lane<16 holds row M=lane, K = {0..7, 16..23};
//                         lane>=16 holds row M=lane-16, K = {8..15, 24..31}.
// B-frag (32x16, 16-bit): lane&15 = column N; lane<16 K = {0..7, 16..23}, lane>=16 K = {8..15, 24..31}.
// C/D   (16x16 f32):      lane&15 = N; VGPR r = M = (lane>>4)*8 + r.
// ---------------------------------------------------------------------------
template<int K, bool SIGMOID, bool HAS_BIAS>
__global__ __launch_bounds__(256) void gemm_bf16_wmma(
    const float* __restrict__ X, const float* __restrict__ W,
    const float* __restrict__ bias, float* __restrict__ Y,
    int M, int N)
{
    const int lane = threadIdx.x & 31;
    const int wave = threadIdx.x >> 5;
    const int nt   = N >> 4;
    const int tile = blockIdx.x * 8 + wave;
    if (tile >= (M >> 4) * nt) return;
    const int tm  = tile / nt;
    const int tn  = tile - tm * nt;
    const int row = lane & 15;
    const int hi  = lane >> 4;

    v8f acc = {0.f, 0.f, 0.f, 0.f, 0.f, 0.f, 0.f, 0.f};

    const float* arow = X + (size_t)(tm * 16 + row) * K + hi * 8;
    const int    bn   = tn * 16 + row;

    #pragma unroll
    for (int kb = 0; kb < K; kb += 32) {
        // speculative prefetch of the next A cacheline
        if (kb + 64 < K) __builtin_prefetch(arow + kb + 64, 0, 0);

        // A fragment: fp32 -> bf16 on the fly
        v4f a0 = *(const v4f*)(arow + kb);
        v4f a1 = *(const v4f*)(arow + kb + 4);
        v4f a2 = *(const v4f*)(arow + kb + 16);
        v4f a3 = *(const v4f*)(arow + kb + 20);
        v16bf a;
        #pragma unroll
        for (int i = 0; i < 4; ++i) {
            a[i]      = (__bf16)a0[i];
            a[4 + i]  = (__bf16)a1[i];
            a[8 + i]  = (__bf16)a2[i];
            a[12 + i] = (__bf16)a3[i];
        }
        // B fragment: column bn, K = 8*hi + {0..7, 16..23} per half-wave (strided gather)
        const float* wcol = W + (size_t)(kb + hi * 8) * N + bn;
        v16bf b;
        #pragma unroll
        for (int e = 0; e < 16; ++e) b[e] = (__bf16)wcol[(size_t)(e + (e & 8)) * N];

        acc = __builtin_amdgcn_wmma_f32_16x16x32_bf16(
            false, a, false, b, (short)0, acc, false, false);
    }

    const int   col = tn * 16 + row;
    const float bv  = HAS_BIAS ? bias[col] : 0.f;
    float* yp = Y + (size_t)(tm * 16 + hi * 8) * N + col;
    #pragma unroll
    for (int r = 0; r < 8; ++r) {
        float v = acc[r] + bv;
        if (SIGMOID) v = fast_rcp(1.f + __expf(-v));
        yp[(size_t)r * N] = v;
    }
}

// ---------------------------------------------------------------------------
// Attention core: one block per (head, b1). Block = 128 threads = 4 waves.
// Each wave processes 4 bands of 16 query rows (4*4*16 = 256 rows).
// HEAD_DIM = 32  ==>  one WMMA per 16x16 logits tile (single K-step).
// ATTN_SCALE is folded into the Q fragment; mask-term + pair bias are
// pre-loaded into the WMMA C operand so the matrix op does the accumulate.
// The sigmoid-gate multiply is fused into the store epilogue.
// ---------------------------------------------------------------------------
__global__ __launch_bounds__(128) void attn_kernel(
    const float* __restrict__ qkv,    // [128, 256, 768] : q|k|v per row
    const float* __restrict__ mask,   // [128, 256]
    const float* __restrict__ pbias,  // [256, 256, 8]
    const float* __restrict__ gate,   // [128, 256, 256] sigmoid(gate) values
    float* __restrict__ wa)           // [128, 256, 256] gated output
{
    __shared__ __bf16 vt[32 * 256];        // V^T: [d=32][m=256], 16 KB
    __shared__ __bf16 pband[4 * 16 * 256]; // per-wave P bands, 32 KB

    const int h    = blockIdx.x;
    const int b1   = blockIdx.y;
    const int tid  = threadIdx.x;
    const int lane = tid & 31;
    const int wave = tid >> 5;
    const int row  = lane & 15;
    const int hi   = lane >> 4;
    const size_t qbase = (size_t)b1 * 256 * 768;

    // Stage V^T into LDS as bf16 (coalesced global reads over d).
    for (int idx = tid; idx < 32 * 256; idx += 128) {
        const int d = idx & 31;
        const int m = idx >> 5;
        vt[d * 256 + m] = (__bf16)qkv[qbase + (size_t)m * 768 + 512 + h * 32 + d];
    }
    __syncthreads();

    __bf16* pb = pband + wave * 16 * 256;   // wave-private P band [16][256]

    for (int rep = 0; rep < 4; ++rep) {
        const int n0 = (wave * 4 + rep) * 16;

        // Q A-fragment, pre-scaled by 1/sqrt(d) before bf16 conversion
        const float* qrow = qkv + qbase + (size_t)(n0 + row) * 768 + h * 32 + hi * 8;
        v16bf aq;
        #pragma unroll
        for (int i = 0; i < 8; ++i) {
            aq[i]     = (__bf16)(qrow[i]      * ATTN_SCALE);
            aq[8 + i] = (__bf16)(qrow[16 + i] * ATTN_SCALE);
        }

        // ---- logits band: 16 tiles of 16x16, one WMMA each,
        //      C operand pre-loaded with mask-term + pair bias ----
        v8f p[16];
        #pragma unroll
        for (int mt = 0; mt < 16; ++mt) {
            // K-matrix rows are contiguous in qkv -> contiguous B frag loads
            const float* krow =
                qkv + qbase + (size_t)(mt * 16 + row) * 768 + 256 + h * 32 + hi * 8;
            v4f k0 = *(const v4f*)(krow);
            v4f k1 = *(const v4f*)(krow + 4);
            v4f k2 = *(const v4f*)(krow + 16);
            v4f k3 = *(const v4f*)(krow + 20);
            v16bf bk;
            #pragma unroll
            for (int i = 0; i < 4; ++i) {
                bk[i]      = (__bf16)k0[i];
                bk[4 + i]  = (__bf16)k1[i];
                bk[8 + i]  = (__bf16)k2[i];
                bk[12 + i] = (__bf16)k3[i];
            }
            const int   mcol  = mt * 16 + row;
            const float mterm = (mask[b1 * 256 + mcol] - 1.f) * MASK_NEG;
            const float* bp   = pbias + ((size_t)(n0 + hi * 8) * 256 + mcol) * 8 + h;
            v8f cinit;
            #pragma unroll
            for (int r = 0; r < 8; ++r) cinit[r] = mterm + bp[(size_t)r * 256 * 8];

            p[mt] = __builtin_amdgcn_wmma_f32_16x16x32_bf16(
                false, aq, false, bk, (short)0, cinit, false, false);
        }

        // ---- softmax: each row lives in one 16-lane half-wave ----
        #pragma unroll
        for (int r = 0; r < 8; ++r) {
            float mx = -3.0e38f;
            #pragma unroll
            for (int mt = 0; mt < 16; ++mt) mx = fmaxf(mx, p[mt][r]);
            #pragma unroll
            for (int s = 1; s < 16; s <<= 1) mx = fmaxf(mx, __shfl_xor(mx, s, 32));
            float sum = 0.f;
            #pragma unroll
            for (int mt = 0; mt < 16; ++mt) {
                const float e = __expf(p[mt][r] - mx);
                p[mt][r] = e;
                sum += e;
            }
            #pragma unroll
            for (int s = 1; s < 16; s <<= 1) sum += __shfl_xor(sum, s, 32);
            const float inv = fast_rcp(sum);
            #pragma unroll
            for (int mt = 0; mt < 16; ++mt) p[mt][r] *= inv;
        }

        // ---- C-layout -> A-layout transpose via wave-private LDS band ----
        #pragma unroll
        for (int mt = 0; mt < 16; ++mt) {
            #pragma unroll
            for (int r = 0; r < 8; ++r)
                pb[(hi * 8 + r) * 256 + mt * 16 + row] = (__bf16)p[mt][r];
        }

        // ---- wa band = P[16x256] x V[256x32] : 2 x 8 WMMAs ----
        #pragma unroll
        for (int dt = 0; dt < 2; ++dt) {
            v8f acc = {0.f, 0.f, 0.f, 0.f, 0.f, 0.f, 0.f, 0.f};
            #pragma unroll
            for (int kb = 0; kb < 8; ++kb) {
                const __bf16* prow = pb + row * 256 + kb * 32 + hi * 16;
                v8bf plo = *(const v8bf*)prow;
                v8bf phv = *(const v8bf*)(prow + 8);
                v16bf ap;
                #pragma unroll
                for (int i = 0; i < 8; ++i) { ap[i] = plo[i]; ap[8 + i] = phv[i]; }

                const __bf16* vrow = vt + (dt * 16 + row) * 256 + kb * 32 + hi * 16;
                v16bf bv = *(const v16bf*)vrow;   // 32B aligned LDS load

                acc = __builtin_amdgcn_wmma_f32_16x16x32_bf16(
                    false, ap, false, bv, (short)0, acc, false, false);
            }
            // fused sigmoid-gate multiply + store
            #pragma unroll
            for (int r = 0; r < 8; ++r) {
                const int nrow = n0 + hi * 8 + r;
                const size_t oidx =
                    ((size_t)b1 * 256 + nrow) * 256 + h * 32 + dt * 16 + row;
                wa[oidx] = acc[r] * gate[oidx];
            }
        }
    }
}

// ---------------------------------------------------------------------------
extern "C" void kernel_launch(void* const* d_in, const int* in_sizes, int n_in,
                              void* d_out, int out_size, void* d_ws, size_t ws_size,
                              hipStream_t stream)
{
    (void)in_sizes; (void)n_in; (void)out_size; (void)ws_size;

    const float* in_data = (const float*)d_in[0];  // [1,128,256,256]
    const float* mask    = (const float*)d_in[1];  // [1,128,1,1,256]
    const float* nb_bias = (const float*)d_in[2];  // [1,256,256,8]
    const float* W_qkv   = (const float*)d_in[3];  // [256,768]
    const float* W_gate  = (const float*)d_in[4];  // [256,256]
    const float* g_bias  = (const float*)d_in[5];  // [256]
    const float* W_o     = (const float*)d_in[6];  // [256,256]
    const float* b_o     = (const float*)d_in[7];  // [256]
    float* out = (float*)d_out;

    const int C = 256;
    const int Mrows = 128 * 256;  // 32768

    float* qkv  = (float*)d_ws;                    // [32768, 768]
    float* gate = qkv  + (size_t)Mrows * 768;      // [32768, 256]
    float* wa   = gate + (size_t)Mrows * C;        // [32768, 256]

    // 1) fused QKV projection
    gemm_bf16_wmma<256, false, false>
        <<<(Mrows / 16) * (768 / 16) / 8, 256, 0, stream>>>(
            in_data, W_qkv, nullptr, qkv, Mrows, 768);

    // 2) gate projection with fused sigmoid(x + gating_bias)
    gemm_bf16_wmma<256, true, true>
        <<<(Mrows / 16) * (C / 16) / 8, 256, 0, stream>>>(
            in_data, W_gate, g_bias, gate, Mrows, C);

    // 3) attention: logits -> softmax -> P*V, gated store (per batch, per head)
    dim3 agrid(8, 128);
    attn_kernel<<<agrid, 128, 0, stream>>>(qkv, mask, nb_bias, gate, wa);

    // 4) output projection with fused +b_o
    gemm_bf16_wmma<256, false, true>
        <<<(Mrows / 16) * (C / 16) / 8, 256, 0, stream>>>(
            wa, W_o, b_o, out, Mrows, C);
}
